// WENONetwork_28630251995291
// MI455X (gfx1250) — matrix operand order repaired and checked by reading, not compiled, not run on hardware
//
#include <hip/hip_runtime.h>

typedef __attribute__((ext_vector_type(16))) _Float16 v16h;
typedef __attribute__((ext_vector_type(8)))  _Float16 v8h;
typedef __attribute__((ext_vector_type(8)))  float    v8f;
typedef __attribute__((ext_vector_type(4)))  unsigned int u32x4;
typedef __attribute__((ext_vector_type(8)))  int      i32x8;
typedef __attribute__((ext_vector_type(4)))  int      i32x4;

#define TILE    64            // output points per workgroup
#define EPOS    80            // extended positions (TILE + 2*8 halo) = 5 N-tiles of 16
#define APAD    2             // zero-padded rows each end of activation tile
#define ROWS    (EPOS + 2*APAD)   // 84 rows in LDS activation buffers
#define UH      10            // uu halo each side
#define UE      84            // uu tile size (TILE + 2*UH)
#define CHPAD   96            // channel padding (covers max 80 ch, multiple of 32)
#define NT      5             // EPOS / 16
#define NWAVES  8             // 256 threads / wave32

// packed-weight offsets (units: _Float16). frag = 512 halfs (16x32 A tile).
// L2: MT=3 taps=5 KC=1 -> 15 frags ; L3: 5*1*2=10 ; L4: 3*1*3=9 ; L5: 2*3*2=12 ; L6: 1
#define OFF2 0
#define OFF3 (15 * 512)
#define OFF4 (25 * 512)
#define OFF5 (34 * 512)
#define OFF6 (46 * 512)
#define WTOTAL (47 * 512)     // 24064 halfs = 48128 bytes

__device__ __forceinline__ float elu_f(float x) {
    return x > 0.f ? x : (__expf(x) - 1.f);
}

// ---------------------------------------------------------------------------
// Pack conv weights (Cout,Cin,kk) f32 -> f16 A-fragments in the CDNA5 16-bit
// A-matrix VGPR layout: lane L holds M = L%16; half h -> K in this code's contiguous order (not ISA table order):
//   g=L/16, K = g*16 + h (contiguous, matches B)
// Fragment order: ((mt*kk + t)*KC + kc), 512 halfs each, per-lane contiguous.
// ---------------------------------------------------------------------------
__global__ void pack_weights(const float* __restrict__ w, _Float16* __restrict__ dst,
                             int Cout, int Cin, int kk, int MT, int KC)
{
    int total = MT * kk * KC * 512;
    for (int i = blockIdx.x * blockDim.x + threadIdx.x; i < total;
         i += gridDim.x * blockDim.x) {
        int h    = i & 15;
        int lane = (i >> 4) & 31;
        int frag = i >> 9;
        int kc   = frag % KC;
        int t    = (frag / KC) % kk;
        int mt   = frag / (KC * kk);
        int m    = mt * 16 + (lane & 15);
        int g    = lane >> 4;
        int v    = h >> 1;
        int pb   = h & 1;
        int k    = g * 16 + h; (void)v; (void)pb;
        int cin  = kc * 32 + k;
        float val = (m < Cout && cin < Cin) ? w[(m * Cin + cin) * kk + t] : 0.f;
        dst[i] = (_Float16)val;
    }
}

// ---------------------------------------------------------------------------
// One WMMA conv layer over the LDS-resident activation tile.
// in/out: [ROWS][CHPAD] f16, rows [0,APAD) and [ROWS-APAD,ROWS) are zero,
// logical position e lives in row e+APAD. Padding channels are zero.
// D(16x16) = sum_t sum_kc A(w,LDS) x B(act shifted by tap) + bias, then ELU.
// B layout per lane: col n = L%16, half h -> row K = (L/16)*16 + h
//   -> 16 contiguous halfs in LDS ([pos][ch], K = channel) = 2x ds_load_b128,
//   tap/kc offsets fold into ds offset immediates (no per-tap VALU).
// ---------------------------------------------------------------------------
template <int COUT, int MT, int TAPS, int PAD, int KC>
__device__ __forceinline__ void conv_layer(const _Float16* __restrict__ in,
                                           _Float16* __restrict__ out,
                                           const _Float16* __restrict__ wf,
                                           const float* __restrict__ bias,
                                           int wave, int lane)
{
    const int n = lane & 15, g = lane >> 4;
    for (int task = wave; task < MT * NT; task += NWAVES) {   // wave-uniform loop
        const int mt = task / NT, nt = task % NT;
        v8f c;
#pragma unroll
        for (int r = 0; r < 8; ++r) {
            int ch = mt * 16 + g * 8 + r;                     // C/D row M = 8*g + r
            c[r] = (ch < COUT) ? bias[ch] : 0.f;
        }
        // B base row for tap 0: logical (nt*16 + n - PAD) -> +APAD in buffer
        const _Float16* bbase = in + (nt * 16 + n + APAD - PAD) * CHPAD + g * 16;
        const _Float16* abase = wf + (mt * TAPS * KC * 32 + lane) * 16;
#pragma unroll
        for (int t = 0; t < TAPS; ++t) {
#pragma unroll
            for (int kc = 0; kc < KC; ++kc) {
                v16h a = *(const v16h*)(abase + (t * KC + kc) * 512);
                v16h b = *(const v16h*)(bbase + t * CHPAD + kc * 32);
                c = __builtin_amdgcn_wmma_f32_16x16x32_f16(
                        false, a, false, b, (short)0, c, false, false);
            }
        }
        v8h o;
#pragma unroll
        for (int r = 0; r < 8; ++r) o[r] = (_Float16)elu_f(c[r]);
        *(v8h*)(out + (nt * 16 + n + APAD) * CHPAD + mt * 16 + g * 8) = o;
    }
}

// ---------------------------------------------------------------------------
// Fused kernel: TDM weight stage -> uu tile -> central diff -> 6-layer CNN
// (WMMA, LDS-resident) -> bm -> WENO5.
// ---------------------------------------------------------------------------
__global__ __launch_bounds__(256) void weno_cnn_kernel(
    const float* __restrict__ uu, int N,
    const float* __restrict__ w1, const float* __restrict__ b1,
    const float* __restrict__ b2, const float* __restrict__ b3,
    const float* __restrict__ b4, const float* __restrict__ b5,
    const float* __restrict__ b6,
    const _Float16* __restrict__ wp,
    float* __restrict__ out)
{
    __shared__ __attribute__((aligned(64))) _Float16 actA[ROWS * CHPAD];
    __shared__ __attribute__((aligned(64))) _Float16 actB[ROWS * CHPAD];
    __shared__ __attribute__((aligned(64))) _Float16 wS[WTOTAL];
    __shared__ float uuS[UE];
    __shared__ float difS[ROWS];
    __shared__ float bmS[EPOS];

    const int tid  = threadIdx.x;
    const int wave = tid >> 5, lane = tid & 31;
    const long gbase = (long)blockIdx.x * TILE;

    // ---- Tensor Data Mover: DMA all packed weights (48128 B) global -> LDS.
    // 1-D tile descriptor: data_size=2B, dim0 = WTOTAL elements. Issued once
    // per workgroup by wave 0; overlaps with uu load / zero-fill / layer 1.
    if (wave == 0) {
        unsigned long long ga = (unsigned long long)(const void*)wp;
        unsigned ldsoff = (unsigned)(unsigned long long)(const void*)&wS[0];
        u32x4 g0;
        g0.x = 1u;                                            // count=1 (valid, user)
        g0.y = ldsoff;                                        // lds_addr
        g0.z = (unsigned)(ga & 0xFFFFFFFFu);                  // global_addr[31:0]
        g0.w = (unsigned)((ga >> 32) & 0x01FFFFFFu) | (2u << 30);  // addr[56:32] | type=2
        i32x8 g1;
        g1[0] = 0x00010000;                                   // data_size=1 (2 bytes)
        g1[1] = (int)((WTOTAL & 0xFFFF) << 16);               // tensor_dim0[15:0]
        g1[2] = (int)((WTOTAL >> 16) | (1 << 16));            // dim0[31:16] | tensor_dim1=1
        g1[3] = (int)((WTOTAL & 0xFFFF) << 16);               // tile_dim0 = WTOTAL
        g1[4] = 1;                                            // tile_dim1=1, tile_dim2=0
        g1[5] = WTOTAL;                                       // tensor_dim0_stride[31:0]
        g1[6] = 0;
        g1[7] = 0;
        i32x4 z4 = {0, 0, 0, 0};
#if defined(__clang_major__) && (__clang_major__ >= 23)
        i32x8 z8 = {0, 0, 0, 0, 0, 0, 0, 0};
        __builtin_amdgcn_tensor_load_to_lds(g0, g1, z4, z4, z8, 0);
#else
        __builtin_amdgcn_tensor_load_to_lds(g0, g1, z4, z4, 0);
#endif
    }

    // uu tile + halo (periodic wrap: matches jnp.roll semantics)
    for (int i = tid; i < UE; i += 256) {
        long g = gbase + i - UH;
        long m = g % N; if (m < 0) m += N;
        uuS[i] = uu[m];
    }
    // zero both activation buffers (pad rows + pad channels must be exactly 0)
    for (int i = tid; i < ROWS * CHPAD / 8; i += 256) {
        v8h z = {};
        ((v8h*)actA)[i] = z;
        ((v8h*)actB)[i] = z;
    }
    __syncthreads();

    // avg_diff == central difference in the interior; zero-padded ends
    for (int i = tid; i < ROWS; i += 256)
        difS[i] = (i >= APAD && i < EPOS + APAD)
                      ? 0.5f * (uuS[i + 1] - uuS[i - 1])   // logical e = i - APAD
                      : 0.f;
    __syncthreads();

    // Layer 1 (1 -> 20, k=5): tiny, plain VALU, no clamps thanks to padding
    for (int idx = tid; idx < 20 * EPOS; idx += 256) {
        int c = idx / EPOS, e = idx % EPOS;
        float acc = b1[c];
#pragma unroll
        for (int t = 0; t < 5; ++t)
            acc += w1[c * 5 + t] * difS[e + t];              // rows e-2..e+2 (+APAD)
        actA[(e + APAD) * CHPAD + c] = (_Float16)elu_f(acc);
    }
    if (wave == 0) __builtin_amdgcn_s_wait_tensorcnt((short)0);  // weights landed
    __syncthreads();

    conv_layer<40, 3, 5, 2, 1>(actA, actB, wS + OFF2, b2, wave, lane); __syncthreads();
    conv_layer<80, 5, 1, 0, 2>(actB, actA, wS + OFF3, b3, wave, lane); __syncthreads();
    conv_layer<40, 3, 1, 0, 3>(actA, actB, wS + OFF4, b4, wave, lane); __syncthreads();
    conv_layer<20, 2, 3, 1, 2>(actB, actA, wS + OFF5, b5, wave, lane); __syncthreads();

    // Layer 6 (20 -> 1, k=1): WMMA + sigmoid + MULT_BIAS -> bmS
    {
        const int n = lane & 15, g = lane >> 4;
        for (int nt = wave; nt < NT; nt += NWAVES) {
            v8f c = {};
            c[0] = (g == 0) ? b6[0] : 0.f;                   // only row M=0 is real
            int p = nt * 16 + n;
            v16h a = *(const v16h*)(wS + OFF6 + lane * 16);
            v16h b = *(const v16h*)(actA + (p + APAD) * CHPAD + g * 16);
            c = __builtin_amdgcn_wmma_f32_16x16x32_f16(
                    false, a, false, b, (short)0, c, false, false);
            if (g == 0) bmS[p] = 1.f / (1.f + __expf(-c[0])) + 0.1f;
        }
    }
    __syncthreads();

    // WENO5 tail in f32 (cheap; epsilon-sensitive)
    if (tid < TILE) {
        const int p = tid;
        const float* U = &uuS[p + UH];
        float umm = U[-2], um = U[-1], u = U[0], up = U[1], upp = U[2], uppp = U[3];

        float fn0 = (11.f * u - 7.f * up + 2.f * upp) * (1.f / 6.f);
        float fn1 = (2.f * um + 5.f * u - up) * (1.f / 6.f);
        float fn2 = (-umm + 5.f * um + 2.f * u) * (1.f / 6.f);
        float fp0 = (11.f * up - 7.f * upp + 2.f * uppp) * (1.f / 6.f);
        float fp1 = (2.f * u + 5.f * up - upp) * (1.f / 6.f);
        float fp2 = (-um + 5.f * u + 2.f * up) * (1.f / 6.f);

        float s;
        s = u - 2.f * up + upp;      float bn0 = (13.f/12.f)*s*s;
        s = 3.f*u - 4.f*up + upp;    bn0 += 0.25f*s*s;
        s = um - 2.f*u + up;         float bn1 = (13.f/12.f)*s*s;
        s = um - up;                 bn1 += 0.25f*s*s;
        s = umm - 2.f*um + u;        float bn2 = (13.f/12.f)*s*s;
        s = umm - 4.f*um + 3.f*u;    bn2 += 0.25f*s*s;
        s = up - 2.f*upp + uppp;     float bp0 = (13.f/12.f)*s*s;
        s = 3.f*up - 4.f*upp + uppp; bp0 += 0.25f*s*s;
        s = u - 2.f*up + upp;        float bp1 = (13.f/12.f)*s*s;
        s = u - upp;                 bp1 += 0.25f*s*s;
        s = um - 2.f*u + up;         float bp2 = (13.f/12.f)*s*s;
        s = um - 4.f*u + 3.f*up;     bp2 += 0.25f*s*s;

        // beta_k *= roll(bm, k-1)[p] = bm[p+1-k]
        float bm0 = bmS[p + 9], bm1 = bmS[p + 8], bm2 = bmS[p + 7];
        bn0 *= bm0; bn1 *= bm1; bn2 *= bm2;
        bp0 *= bm0; bp1 *= bm1; bp2 *= bm2;

        float fluxp, fluxn;
        {
            float brs = (bp2 - bp0) * (bp2 - bp0);
            float e0 = (1e-13f + bp0) * (1e-13f + bp0);
            float e1 = (1e-13f + bp1) * (1e-13f + bp1);
            float e2 = (1e-13f + bp2) * (1e-13f + bp2);
            float o0 = 0.1f / e0 * (brs + e0);
            float o1 = 0.6f / e1 * (brs + e1);
            float o2 = 0.3f / e2 * (brs + e2);
            float inv = 1.f / (o0 + o1 + o2);
            fluxp = (o0 * fp0 + o1 * fp1 + o2 * fp2) * inv;
        }
        {
            float brs = (bn2 - bn0) * (bn2 - bn0);
            float e0 = (1e-13f + bn0) * (1e-13f + bn0);
            float e1 = (1e-13f + bn1) * (1e-13f + bn1);
            float e2 = (1e-13f + bn2) * (1e-13f + bn2);
            float o0 = 0.1f / e0 * (brs + e0);
            float o1 = 0.6f / e1 * (brs + e1);
            float o2 = 0.3f / e2 * (brs + e2);
            float inv = 1.f / (o0 + o1 + o2);
            fluxn = (o0 * fn0 + o1 * fn1 + o2 * fn2) * inv;
        }
        long go = gbase + p;
        if (go < N) out[go] = fluxp - fluxn;
    }
}

extern "C" void kernel_launch(void* const* d_in, const int* in_sizes, int n_in,
                              void* d_out, int out_size, void* d_ws, size_t ws_size,
                              hipStream_t stream)
{
    (void)n_in; (void)out_size; (void)ws_size;
    const float* uu = (const float*)d_in[0];
    const float* w1 = (const float*)d_in[1];  const float* b1 = (const float*)d_in[2];
    const float* w2 = (const float*)d_in[3];  const float* b2 = (const float*)d_in[4];
    const float* w3 = (const float*)d_in[5];  const float* b3 = (const float*)d_in[6];
    const float* w4 = (const float*)d_in[7];  const float* b4 = (const float*)d_in[8];
    const float* w5 = (const float*)d_in[9];  const float* b5 = (const float*)d_in[10];
    const float* w6 = (const float*)d_in[11]; const float* b6 = (const float*)d_in[12];

    _Float16* wp = (_Float16*)d_ws;   // 47 frags * 512 halfs = 48128 B used

    pack_weights<<<8, 256, 0, stream>>>(w2, wp + OFF2, 40, 20, 5, 3, 1);
    pack_weights<<<8, 256, 0, stream>>>(w3, wp + OFF3, 80, 40, 1, 5, 2);
    pack_weights<<<8, 256, 0, stream>>>(w4, wp + OFF4, 40, 80, 1, 3, 3);
    pack_weights<<<8, 256, 0, stream>>>(w5, wp + OFF5, 20, 40, 3, 2, 2);
    pack_weights<<<8, 256, 0, stream>>>(w6, wp + OFF6, 1, 20, 1, 1, 1);

    int N = in_sizes[0];
    int grid = (N + TILE - 1) / TILE;
    weno_cnn_kernel<<<grid, 256, 0, stream>>>(uu, N, w1, b1, b2, b3, b4, b5, b6,
                                              wp, (float*)d_out);
}
